// GatedWCVAEDecoder_21698174780140
// MI455X (gfx1250) — matrix operand order repaired and checked by reading, not compiled, not run on hardware
//
#include <hip/hip_runtime.h>
#include <hip/hip_bf16.h>

// Problem constants (from reference): B=16384, N=512, L=256, E=8, R=32.
#define BB 16384
#define NN 512
#define LL 256
#define EE 8
#define RR 32
#define ROWS 64            // rows of B per workgroup (4 row-blocks of 16)

typedef __attribute__((ext_vector_type(16))) __bf16 v16bf;
typedef __attribute__((ext_vector_type(8)))  float  v8f;

__device__ __forceinline__ unsigned short f2bf(float f) {
    unsigned u = __builtin_bit_cast(unsigned, f);
    unsigned r = u + 0x7FFFu + ((u >> 16) & 1u);   // round-to-nearest-even
    return (unsigned short)(r >> 16);
}

// ---------------------------------------------------------------------------
// Pack A operand (x) into 16x32 bf16 per-lane layout (contiguous k, this code's order; B matches):
// pA[bt16][kt][lane][16], lane m=lane&15 is row, koff=(lane>=16)?16:0,
// vgpr pair i: K = koff+2i, koff+2i+1
// ---------------------------------------------------------------------------
__global__ void pack_a_kernel(const float* __restrict__ x, unsigned short* __restrict__ pA) {
    int id = blockIdx.x * 256 + threadIdx.x;        // < (B/16)*(N/32)*32 = 524288
    int lane = id & 31;
    int kt   = (id >> 5) & 15;
    int bt   = id >> 9;
    int m    = lane & 15;
    int koff = (lane >> 4) * 16;
    const float* xr = x + (size_t)(bt * 16 + m) * NN + kt * 32;
    unsigned short* dst = pA + (size_t)id * 16;
    #pragma unroll
    for (int i = 0; i < 8; ++i) {
        int kb = koff + 2 * i;
        dst[2 * i]     = f2bf(xr[kb]);
        dst[2 * i + 1] = f2bf(xr[kb + 1]);
    }
}

// ---------------------------------------------------------------------------
// Pack B operand: experts 0..7 from W [E,N,L], expert 8 = Wg [N,L].
// pB[e][nt][kt][lane][16]; col n = nt*16 + (lane&15); koff=(lane>=16)?16:0;
// vgpr j holds K = koff+2j, koff+2j+1.
// ---------------------------------------------------------------------------
__global__ void pack_b_kernel(const float* __restrict__ W, const float* __restrict__ Wg,
                              unsigned short* __restrict__ pB) {
    int id = blockIdx.x * 256 + threadIdx.x;        // < 9*16*16*32 = 73728
    int lane = id & 31;
    int kt   = (id >> 5) & 15;
    int nt   = (id >> 9) & 15;
    int e    = id >> 13;
    int n    = nt * 16 + (lane & 15);
    int koff = (lane >> 4) * 16;
    unsigned short* dst = pB + (size_t)id * 16;
    #pragma unroll
    for (int j = 0; j < 8; ++j) {
        int k = kt * 32 + koff + 2 * j;
        float v0, v1;
        if (e < EE) {
            v0 = W[((size_t)e * NN + k) * LL + n];
            v1 = W[((size_t)e * NN + k + 1) * LL + n];
        } else {
            v0 = Wg[(size_t)k * LL + n];
            v1 = Wg[(size_t)(k + 1) * LL + n];
        }
        dst[2 * j]     = f2bf(v0);
        dst[2 * j + 1] = f2bf(v1);
    }
}

// Pack H [R=32, L=256] (0/1 floats) into 32x8 u32 bitmasks.
__global__ void pack_h_kernel(const float* __restrict__ H, unsigned* __restrict__ Hb) {
    int t = threadIdx.x;            // 256 threads
    int r = t >> 3, w = t & 7;
    unsigned bits = 0;
    #pragma unroll
    for (int i = 0; i < 32; ++i)
        if (H[(size_t)r * LL + w * 32 + i] > 0.5f) bits |= (1u << i);
    Hb[r * 8 + w] = bits;
}

// ---------------------------------------------------------------------------
// 64-row x 256-col GEMM micro-tile for one B slot: each wave owns column
// tiles ct0=wave, ct1=wave+8, and 4 row-blocks. One B fetch feeds 4 WMMAs;
// one A fetch feeds 2 WMMAs (8 independent accumulator chains per wave).
// ---------------------------------------------------------------------------
__device__ __forceinline__ void gemm64(const unsigned short* __restrict__ ldsA,
                                       const unsigned short* __restrict__ pB,
                                       int slot, int lane, int ct0, int ct1,
                                       v8f acc[4][2]) {
    #pragma unroll
    for (int rb = 0; rb < 4; ++rb)
        #pragma unroll
        for (int t = 0; t < 2; ++t)
            acc[rb][t] = (v8f){0.f, 0.f, 0.f, 0.f, 0.f, 0.f, 0.f, 0.f};

    const unsigned short* b0base = pB + ((size_t)(slot * 16 + ct0) * 16) * 512 + lane * 16;
    const unsigned short* b1base = pB + ((size_t)(slot * 16 + ct1) * 16) * 512 + lane * 16;

    #pragma unroll 2
    for (int kt = 0; kt < 16; ++kt) {
        v16bf b0 = *(const v16bf*)(b0base + (size_t)kt * 512);
        v16bf b1 = *(const v16bf*)(b1base + (size_t)kt * 512);
        #pragma unroll
        for (int rb = 0; rb < 4; ++rb) {
            v16bf a = *(const v16bf*)&ldsA[(size_t)rb * 8192 + (kt * 32 + lane) * 16];
            acc[rb][0] = __builtin_amdgcn_wmma_f32_16x16x32_bf16(false, a, false, b0,
                                                                 (short)0, acc[rb][0], false, false);
            acc[rb][1] = __builtin_amdgcn_wmma_f32_16x16x32_bf16(false, a, false, b1,
                                                                 (short)0, acc[rb][1], false, false);
        }
    }
}

// ---------------------------------------------------------------------------
// Main kernel: one workgroup (8 waves) owns 64 rows of B.
//   Phase 1: gating GEMM (slot 8), 8 biased thresholds, CRC, argmin.
//   Phase 2: 8 expert GEMMs, masked CRC, running first-min select.
//   Final:   out = non_zero ? sel : decoded, as 0/1 floats.
// ---------------------------------------------------------------------------
__global__ __launch_bounds__(256) void moe_decode_kernel(
        const unsigned short* __restrict__ pA,
        const unsigned short* __restrict__ pB,
        const float* __restrict__ bias,
        const unsigned* __restrict__ Hb,
        float* __restrict__ out) {

    __shared__ unsigned short ldsA[4 * 16 * 32 * 16];        // 64 KB  A operand tiles
    __shared__ unsigned short dgate[EE][ROWS][16];           // 16 KB  gate bits
    __shared__ unsigned short dcur[ROWS][16];                // current expert bits
    __shared__ unsigned short decoded[ROWS][16];             // gate-selected word
    __shared__ unsigned short bestword[ROWS][16];            // expert-selected word
    __shared__ int crcg[ROWS][EE];
    __shared__ int crccur[ROWS];
    __shared__ int bestcrc[ROWS];
    __shared__ int tdmask[ROWS];
    __shared__ int nonzero[ROWS];

    const int tid  = threadIdx.x;
    const int wave = tid >> 5;
    const int lane = tid & 31;
    const int bt   = blockIdx.x;
    const int ct0  = wave;
    const int ct1  = wave + 8;

    // Stage A operand for this 64-row block into LDS (4096 x uint4 = 64 KB).
    {
        const uint4* src = (const uint4*)(pA + (size_t)bt * (4 * 16 * 32 * 16));
        uint4* dst = (uint4*)ldsA;
        for (int i = tid; i < 4096; i += 256) dst[i] = src[i];
    }
    __syncthreads();

    // ---------------- Phase 1: gating GEMM (pB slot 8) ----------------
    {
        v8f acc[4][2];
        gemm64(ldsA, pB, 8, lane, ct0, ct1, acc);

        // 8 biased sign-thresholds -> bit masks (ballot bit i = lane i:
        // lanes 0-15 = (row r, col i), lanes 16-31 = (row r+8, col i)).
        for (int e = 0; e < EE; ++e) {
            float be = bias[e];
            #pragma unroll
            for (int rb = 0; rb < 4; ++rb) {
                #pragma unroll
                for (int r = 0; r < 8; ++r) {
                    unsigned m0 = __builtin_amdgcn_ballot_w32((acc[rb][0][r] + be) < 0.0f);
                    unsigned m1 = __builtin_amdgcn_ballot_w32((acc[rb][1][r] + be) < 0.0f);
                    if (lane == 0) {
                        int row = rb * 16 + r;
                        dgate[e][row][ct0]     = (unsigned short)(m0 & 0xFFFFu);
                        dgate[e][row + 8][ct0] = (unsigned short)(m0 >> 16);
                        dgate[e][row][ct1]     = (unsigned short)(m1 & 0xFFFFu);
                        dgate[e][row + 8][ct1] = (unsigned short)(m1 >> 16);
                    }
                }
            }
        }
    }
    __syncthreads();

    // CRC of all 8 gate candidates: 512 (row, e) pairs over 256 threads.
    for (int p = tid; p < ROWS * EE; p += 256) {
        int row = p >> 3, e = p & 7;
        unsigned d[8];
        #pragma unroll
        for (int w = 0; w < 8; ++w)
            d[w] = (unsigned)dgate[e][row][2 * w] | ((unsigned)dgate[e][row][2 * w + 1] << 16);
        int crc = 0;
        #pragma unroll 4
        for (int r = 0; r < RR; ++r) {
            unsigned xv = 0;
            #pragma unroll
            for (int w = 0; w < 8; ++w) xv ^= Hb[r * 8 + w] & d[w];
            crc += __popc(xv) & 1;
        }
        crcg[row][e] = crc;
    }
    __syncthreads();

    // Per-row gating reduction: argmin (first), dispatch mask, decoded word.
    if (tid < ROWS) {
        int row = tid;
        int mn = 1 << 30, arg = 0;
        #pragma unroll
        for (int e = 0; e < EE; ++e) {
            int cv = crcg[row][e];
            if (cv < mn) { mn = cv; arg = e; }
        }
        int msk = 0;
        #pragma unroll
        for (int e = 0; e < EE; ++e) if (crcg[row][e] == mn) msk |= (1 << e);
        tdmask[row]  = msk;
        nonzero[row] = (mn != 0);
        bestcrc[row] = 1 << 30;                 // INF sentinel (reference uses 100)
        #pragma unroll
        for (int w = 0; w < 16; ++w) decoded[row][w] = dgate[arg][row][w];
    }
    __syncthreads();

    // ---------------- Phase 2: expert GEMMs with masked dispatch ----------------
    for (int e = 0; e < EE; ++e) {
        {
            v8f acc[4][2];
            gemm64(ldsA, pB, e, lane, ct0, ct1, acc);
            #pragma unroll
            for (int rb = 0; rb < 4; ++rb) {
                #pragma unroll
                for (int r = 0; r < 8; ++r) {
                    unsigned m0 = __builtin_amdgcn_ballot_w32(acc[rb][0][r] < 0.0f);
                    unsigned m1 = __builtin_amdgcn_ballot_w32(acc[rb][1][r] < 0.0f);
                    if (lane == 0) {
                        int row = rb * 16 + r;
                        dcur[row][ct0]     = (unsigned short)(m0 & 0xFFFFu);
                        dcur[row + 8][ct0] = (unsigned short)(m0 >> 16);
                        dcur[row][ct1]     = (unsigned short)(m1 & 0xFFFFu);
                        dcur[row + 8][ct1] = (unsigned short)(m1 >> 16);
                    }
                }
            }
        }
        if (tid < ROWS) crccur[tid] = 0;
        __syncthreads();

        // Parallel masked CRC: 8 threads per row, 4 parity rows each (512 pairs).
        for (int p = tid; p < ROWS * 8; p += 256) {
            int row = p >> 3, grp = p & 7;
            if ((tdmask[row] >> e) & 1) {
                unsigned d[8];
                #pragma unroll
                for (int w = 0; w < 8; ++w)
                    d[w] = (unsigned)dcur[row][2 * w] | ((unsigned)dcur[row][2 * w + 1] << 16);
                int q = 0;
                #pragma unroll
                for (int r = grp * 4; r < grp * 4 + 4; ++r) {
                    unsigned xv = 0;
                    #pragma unroll
                    for (int w = 0; w < 8; ++w) xv ^= Hb[r * 8 + w] & d[w];
                    q += __popc(xv) & 1;
                }
                atomicAdd(&crccur[row], q);
            }
        }
        __syncthreads();

        // Running first-occurrence min (strict < keeps earliest expert on ties).
        if (tid < ROWS) {
            int row = tid;
            if (((tdmask[row] >> e) & 1) && crccur[row] < bestcrc[row]) {
                bestcrc[row] = crccur[row];
                #pragma unroll
                for (int w = 0; w < 16; ++w) bestword[row][w] = dcur[row][w];
            }
        }
        __syncthreads();
    }

    // ---------------- Final select + write 64 rows x 256 cols of 0/1 floats ----
    for (int u = tid; u < ROWS * 16; u += 256) {
        int row = u >> 4, ct = u & 15;
        unsigned short wv = nonzero[row] ? bestword[row][ct] : decoded[row][ct];
        float* o = out + ((size_t)bt * ROWS + row) * LL + ct * 16;
        #pragma unroll
        for (int i = 0; i < 16; ++i) o[i] = (float)((wv >> i) & 1u);
    }
}

// ---------------------------------------------------------------------------
// Launcher. Inputs (setup_inputs order): x[B,N] f32, Wg[N,L] f32,
// state_bias[E] f32, W[E,N,L] f32, H[R,L] f32. Output: [B,L] f32.
// Workspace layout: Hb (1 KB) | packedB (2.25 MB) | packedA (16 MB).
// ---------------------------------------------------------------------------
extern "C" void kernel_launch(void* const* d_in, const int* in_sizes, int n_in,
                              void* d_out, int out_size, void* d_ws, size_t ws_size,
                              hipStream_t stream) {
    const float* x    = (const float*)d_in[0];
    const float* Wg   = (const float*)d_in[1];
    const float* bias = (const float*)d_in[2];
    const float* W    = (const float*)d_in[3];
    const float* H    = (const float*)d_in[4];
    float* out = (float*)d_out;

    char* ws = (char*)d_ws;
    unsigned*       Hb = (unsigned*)ws;                          // 1 KB
    unsigned short* pB = (unsigned short*)(ws + 4096);           // 9*16*16*512 u16 = 2.25 MB
    unsigned short* pA = (unsigned short*)(ws + 4096 + 2359296); // (B/16)*16*32*16 u16 = 16 MB

    pack_h_kernel<<<1, 256, 0, stream>>>(H, Hb);
    pack_b_kernel<<<288, 256, 0, stream>>>(W, Wg, pB);
    pack_a_kernel<<<2048, 256, 0, stream>>>(x, pA);
    moe_decode_kernel<<<BB / ROWS, 256, 0, stream>>>(pA, pB, bias, Hb, out);
    (void)in_sizes; (void)n_in; (void)out_size; (void)ws_size;
}
